// Box3dTransformerEncoderLayer_4638564680381
// MI455X (gfx1250) — compile-verified, matrix operand order checked by reading, not run on hardware
//
#include <hip/hip_runtime.h>

// ---------------------------------------------------------------------------
// Problem constants (from the reference)
// ---------------------------------------------------------------------------
#define BB      2
#define LV_C    21760            // 128*128 + 64*64 + 32*32 + 16*16
#define MT      (BB * LV_C)      // 43520 rows  (= 340 * 128)
#define DMODEL  256
#define NHEAD   8
#define HDIM    32
#define DFF_C   1024

typedef __attribute__((ext_vector_type(16))) __bf16 v16bf;
typedef __attribute__((ext_vector_type(8)))  __bf16 v8bf;
typedef __attribute__((ext_vector_type(8)))  float  v8f;
typedef int v4i __attribute__((vector_size(16)));   // matches builtin param type

#define AS1 __attribute__((address_space(1)))
#define AS3 __attribute__((address_space(3)))

// Direct global->LDS async copy (ASYNCcnt) if the toolchain exposes it.
#if defined(__has_builtin)
#if __has_builtin(__builtin_amdgcn_global_load_async_to_lds_b128)
#define HAVE_ASYNC_LDS 1
#endif
#endif
#ifndef HAVE_ASYNC_LDS
#define HAVE_ASYNC_LDS 0
#endif

static __device__ __forceinline__ void wait_async_lds() {
#if HAVE_ASYNC_LDS
#if __has_builtin(__builtin_amdgcn_s_wait_asynccnt)
  __builtin_amdgcn_s_wait_asynccnt(0);
#else
  asm volatile("s_wait_asynccnt 0x0" ::: "memory");
#endif
#endif
}

// round-to-nearest-even f32 -> bf16 (pure bit ops; no reliance on fptrunc)
static __device__ __forceinline__ __bf16 f2bf(float f) {
  unsigned u = __builtin_bit_cast(unsigned, f);
  unsigned r = (u + 0x7FFFu + ((u >> 16) & 1u)) >> 16;
  unsigned short s = (unsigned short)r;
  return __builtin_bit_cast(__bf16, s);
}

static __device__ __forceinline__ v16bf combine8(v8bf lo, v8bf hi) {
  v16bf r;
#pragma unroll
  for (int i = 0; i < 8; ++i) { r[i] = lo[i]; r[i + 8] = hi[i]; }
  return r;
}

// ---------------------------------------------------------------------------
// Elementwise converters
// ---------------------------------------------------------------------------
__global__ __launch_bounds__(256) void k_addcvt(const float* __restrict__ src,
                                                const float* __restrict__ pos,
                                                __bf16* __restrict__ qbf,
                                                __bf16* __restrict__ srcbf,
                                                int n) {
  int i = blockIdx.x * 256 + threadIdx.x;
  if (i < n) {
    float s = src[i];
    qbf[i]   = f2bf(s + pos[i]);
    srcbf[i] = f2bf(s);
  }
}

__global__ __launch_bounds__(256) void k_cvt(const float* __restrict__ in,
                                             __bf16* __restrict__ out, int n) {
  int i = blockIdx.x * 256 + threadIdx.x;
  if (i < n) out[i] = f2bf(in[i]);
}

// ---------------------------------------------------------------------------
// WMMA GEMM: C[M x N] = A[M x K] * W[N x K]^T + bias
//   block: 256 threads (8 waves), tile 128(M) x 64(N), K step 32
//   double-buffered LDS; async global->LDS staging when available
//   EPI==0: store f32 to Cf ; EPI==1: relu + store bf16 to Cb
// ---------------------------------------------------------------------------
template <int N, int K, int EPI>
__global__ __launch_bounds__(256) void gemm_wmma(const __bf16* __restrict__ A,
                                                 const __bf16* __restrict__ W,
                                                 const float* __restrict__ bias,
                                                 float* __restrict__ Cf,
                                                 __bf16* __restrict__ Cb) {
  constexpr int PITCH = 40;                 // halves; pads 32 -> 40 (bank spread)
  __shared__ __bf16 sA[2][128 * PITCH];     // 2 x 10 KB
  __shared__ __bf16 sB[2][64 * PITCH];      // 2 x  5 KB

  const int tid    = threadIdx.x;
  const int lane   = tid & 31;
  const int wave   = tid >> 5;              // 0..7
  const int l16    = lane & 15;
  const int half   = lane >> 4;             // 0 | 1
  const int khalf  = half * 8;              // k sub-offset for this half-wave
  const int blockM = blockIdx.y * 128;
  const int tileN0 = blockIdx.x * 64;

  // staging roles: 4 threads per row, 16B (8 halves) each
  const int part  = tid & 3;
  const int srow  = tid >> 2;               // 0..63
  const int kpart = part * 8;

  auto stage = [&](int buf, int k0) {
    const __bf16* ga0 = A + (size_t)(blockM + srow) * K + k0 + kpart;
    const __bf16* ga1 = A + (size_t)(blockM + srow + 64) * K + k0 + kpart;
    const __bf16* gb  = W + (size_t)(tileN0 + srow) * K + k0 + kpart;
    __bf16* la0 = &sA[buf][srow * PITCH + kpart];
    __bf16* la1 = &sA[buf][(srow + 64) * PITCH + kpart];
    __bf16* lb  = &sB[buf][srow * PITCH + kpart];
#if HAVE_ASYNC_LDS
    __builtin_amdgcn_global_load_async_to_lds_b128((AS1 v4i*)ga0,
                                                   (AS3 v4i*)la0, 0, 0);
    __builtin_amdgcn_global_load_async_to_lds_b128((AS1 v4i*)ga1,
                                                   (AS3 v4i*)la1, 0, 0);
    __builtin_amdgcn_global_load_async_to_lds_b128((AS1 v4i*)gb,
                                                   (AS3 v4i*)lb, 0, 0);
#else
    *(v8bf*)la0 = *(const v8bf*)ga0;
    *(v8bf*)la1 = *(const v8bf*)ga1;
    *(v8bf*)lb  = *(const v8bf*)gb;
#endif
  };

  v8f acc[4];
#pragma unroll
  for (int t = 0; t < 4; ++t) acc[t] = v8f{};

  // prologue: fill buffer 0
  stage(0, 0);
  wait_async_lds();
  __syncthreads();

  for (int k0 = 0; k0 < K; k0 += 32) {
    const int cur = (k0 >> 5) & 1;
    const bool more = (k0 + 32) < K;
    if (more) stage(cur ^ 1, k0 + 32);      // overlap next stage with compute

    // ---- A fragment: lane<16 -> k {0..7,16..23}; lane>=16 -> {8..15,24..31}
    const int arow = wave * 16 + l16;
    v16bf af = combine8(*(const v8bf*)&sA[cur][arow * PITCH + khalf],
                        *(const v8bf*)&sA[cur][arow * PITCH + khalf + 16]);
#pragma unroll
    for (int t = 0; t < 4; ++t) {
      const int brow = t * 16 + l16;
      v16bf bf = combine8(*(const v8bf*)&sB[cur][brow * PITCH + khalf],
                          *(const v8bf*)&sB[cur][brow * PITCH + khalf + 16]);
      acc[t] = __builtin_amdgcn_wmma_f32_16x16x32_bf16(
          false, af, false, bf, (short)0, acc[t], false, false);
    }

    if (more) {
      wait_async_lds();                     // our stage writes have landed
      __syncthreads();                      // everyone done reading `cur`
    }
  }

  // ---- epilogue: C layout lane=N col, vgpr r -> row r (+8 for upper half)
  const int rowAdd = half * 8;
#pragma unroll
  for (int t = 0; t < 4; ++t) {
    const int n  = tileN0 + t * 16 + l16;
    const float bv = bias[n];
#pragma unroll
    for (int r = 0; r < 8; ++r) {
      const int m = blockM + wave * 16 + r + rowAdd;
      float val = acc[t][r] + bv;
      if constexpr (EPI == 1) {
        val = fmaxf(val, 0.f);
        Cb[(size_t)m * N + n] = f2bf(val);
      } else {
        Cf[(size_t)m * N + n] = val;
      }
    }
  }
}

// ---------------------------------------------------------------------------
// Box-attention sampling: block = one (b,q); thread = (head h, dim d)
// ---------------------------------------------------------------------------
__global__ __launch_bounds__(256) void k_sample(const float* __restrict__ vproj,
                                                const float* __restrict__ awb,
                                                const float* __restrict__ obb,
                                                const float* __restrict__ rw,
                                                __bf16* __restrict__ outbf) {
  const int bq = blockIdx.x;                 // 0..MT-1
  const int d  = threadIdx.x & 31;
  const int h  = threadIdx.x >> 5;
  const int b  = bq / LV_C;

  const float* r = rw + (size_t)bq * 7;
  const float cx = r[0], cy = r[1], sw = r[3], sh = r[4], ang = r[6];
  const float cA = __cosf(ang), sA = __sinf(ang);

  // softmax over the 16 (level,point) logits of this head
  const float* awp = awb + ((size_t)bq * NHEAD + h) * 16;
  float p16[16];
  float mx = -1e30f;
#pragma unroll
  for (int i = 0; i < 16; ++i) { p16[i] = awp[i]; mx = fmaxf(mx, p16[i]); }
  float ssum = 0.f;
#pragma unroll
  for (int i = 0; i < 16; ++i) { p16[i] = __expf(p16[i] - mx); ssum += p16[i]; }
  const float inv = 1.f / ssum;
#pragma unroll
  for (int i = 0; i < 16; ++i) p16[i] *= inv;

  const float* obp = obb + ((size_t)bq * NHEAD + h) * 16;

  constexpr int   dims[4]   = {128, 64, 32, 16};
  constexpr int   starts[4] = {0, 16384, 20480, 21504};
  constexpr float kerx[4]   = {-0.25f, 0.25f, -0.25f, 0.25f};
  constexpr float kery[4]   = {-0.25f, -0.25f, 0.25f, 0.25f};

  float acc = 0.f;
#pragma unroll
  for (int lvl = 0; lvl < 4; ++lvl) {
    const float o0 = obp[lvl * 4 + 0], o1 = obp[lvl * 4 + 1];
    const float o2 = obp[lvl * 4 + 2], o3 = obp[lvl * 4 + 3];
    const float bx = cx + o0 * 0.125f * sw;
    const float by = cy + o1 * 0.125f * sh;
    const float rsw = fmaxf(sw + o2 * 0.125f * sw, 0.f);
    const float rsh = fmaxf(sh + o3 * 0.125f * sh, 0.f);
    const int   Wd = dims[lvl], Hd = dims[lvl];
    const float* vbase =
        vproj + ((size_t)(b * LV_C + starts[lvl])) * DMODEL + h * HDIM + d;
#pragma unroll
    for (int p = 0; p < 4; ++p) {
      const float gx0 = kerx[p] * rsw, gy0 = kery[p] * rsh;
      const float gx = bx + cA * gx0 - sA * gy0;
      const float gy = by + sA * gx0 + cA * gy0;
      const float xx = gx * (float)Wd - 0.5f;
      const float yy = gy * (float)Hd - 0.5f;
      const float x0f = floorf(xx), y0f = floorf(yy);
      const float wx = xx - x0f, wy = yy - y0f;
      const int x0 = (int)x0f, y0 = (int)y0f;
      const float attp = p16[lvl * 4 + p];
      const float cw[4] = {(1.f - wx) * (1.f - wy), wx * (1.f - wy),
                           (1.f - wx) * wy,         wx * wy};
      const int cxs[4] = {x0, x0 + 1, x0, x0 + 1};
      const int cys[4] = {y0, y0, y0 + 1, y0 + 1};
#pragma unroll
      for (int c = 0; c < 4; ++c) {
        const int xi = cxs[c], yi = cys[c];
        if (xi >= 0 && xi < Wd && yi >= 0 && yi < Hd)
          acc += attp * cw[c] * vbase[(size_t)(yi * Wd + xi) * DMODEL];
      }
    }
  }
  outbf[(size_t)bq * DMODEL + h * HDIM + d] = f2bf(acc);
}

// ---------------------------------------------------------------------------
// Residual + LayerNorm: wave per row (8 rows/block), wave32 shfl reductions
// ---------------------------------------------------------------------------
__global__ __launch_bounds__(256) void k_ln(const float* __restrict__ a,
                                            const float* __restrict__ res,
                                            const float* __restrict__ w,
                                            const float* __restrict__ bvec,
                                            float* __restrict__ yf,
                                            __bf16* __restrict__ ybf,
                                            int rows) {
  const int wave = threadIdx.x >> 5;
  const int lane = threadIdx.x & 31;
  const int row  = blockIdx.x * 8 + wave;
  if (row >= rows) return;

  const float* ap = a + (size_t)row * DMODEL;
  const float* rp = res + (size_t)row * DMODEL;
  float x[8];
  float s = 0.f;
#pragma unroll
  for (int i = 0; i < 8; ++i) {
    x[i] = ap[lane + i * 32] + rp[lane + i * 32];
    s += x[i];
  }
#pragma unroll
  for (int off = 16; off >= 1; off >>= 1) s += __shfl_xor(s, off, 32);
  const float mean = s * (1.f / 256.f);

  float v = 0.f;
#pragma unroll
  for (int i = 0; i < 8; ++i) { const float dd = x[i] - mean; v += dd * dd; }
#pragma unroll
  for (int off = 16; off >= 1; off >>= 1) v += __shfl_xor(v, off, 32);
  const float rstd = rsqrtf(v * (1.f / 256.f) + 1e-5f);

#pragma unroll
  for (int i = 0; i < 8; ++i) {
    const int e = lane + i * 32;
    const float y = (x[i] - mean) * rstd * w[e] + bvec[e];
    yf[(size_t)row * DMODEL + e] = y;
    if (ybf) ybf[(size_t)row * DMODEL + e] = f2bf(y);
  }
}

// ---------------------------------------------------------------------------
// Launch
// ---------------------------------------------------------------------------
extern "C" void kernel_launch(void* const* d_in, const int* in_sizes, int n_in,
                              void* d_out, int out_size, void* d_ws,
                              size_t ws_size, hipStream_t stream) {
  (void)in_sizes; (void)n_in; (void)out_size; (void)ws_size;

  const float* src   = (const float*)d_in[0];
  const float* pos   = (const float*)d_in[1];
  const float* rw    = (const float*)d_in[4];
  const float* vpw   = (const float*)d_in[5];
  const float* vpb   = (const float*)d_in[6];
  const float* opw   = (const float*)d_in[7];
  const float* opb   = (const float*)d_in[8];
  const float* boxw  = (const float*)d_in[9];
  const float* boxb  = (const float*)d_in[10];
  const float* attw  = (const float*)d_in[11];
  const float* attb  = (const float*)d_in[12];
  const float* l1w   = (const float*)d_in[13];
  const float* l1b   = (const float*)d_in[14];
  const float* l2w   = (const float*)d_in[15];
  const float* l2b   = (const float*)d_in[16];
  const float* ln1w  = (const float*)d_in[17];
  const float* ln1b  = (const float*)d_in[18];
  const float* ln2w  = (const float*)d_in[19];
  const float* ln2b  = (const float*)d_in[20];

  size_t off = 0;
  auto carve = [&](size_t bytes) -> void* {
    void* p = (char*)d_ws + off;
    off += (bytes + 255) & ~(size_t)255;
    return p;
  };

  __bf16* qbf    = (__bf16*)carve((size_t)MT * DMODEL * 2);
  __bf16* srcbf  = (__bf16*)carve((size_t)MT * DMODEL * 2);
  __bf16* outbf  = (__bf16*)carve((size_t)MT * DMODEL * 2);
  __bf16* xbf    = (__bf16*)carve((size_t)MT * DMODEL * 2);
  __bf16* hbf    = (__bf16*)carve((size_t)MT * DFF_C * 2);
  float*  vbuf   = (float*)carve((size_t)MT * DMODEL * 4);
  float*  awbuf  = (float*)carve((size_t)MT * 128 * 4);
  float*  obbuf  = (float*)carve((size_t)MT * 128 * 4);
  float*  src2   = (float*)carve((size_t)MT * DMODEL * 4);
  float*  x1     = (float*)carve((size_t)MT * DMODEL * 4);
  float*  ffn    = (float*)carve((size_t)MT * DMODEL * 4);
  __bf16* vpwb   = (__bf16*)carve(65536 * 2);
  __bf16* opwb   = (__bf16*)carve(65536 * 2);
  __bf16* attwb  = (__bf16*)carve(32768 * 2);
  __bf16* boxwb  = (__bf16*)carve(32768 * 2);
  __bf16* l1wb   = (__bf16*)carve(262144 * 2);
  __bf16* l2wb   = (__bf16*)carve(262144 * 2);

  const int nAct = MT * DMODEL;                        // 11,141,120
  k_addcvt<<<nAct / 256, 256, 0, stream>>>(src, pos, qbf, srcbf, nAct);
  k_cvt<<<65536 / 256, 256, 0, stream>>>(vpw, vpwb, 65536);
  k_cvt<<<65536 / 256, 256, 0, stream>>>(opw, opwb, 65536);
  k_cvt<<<32768 / 256, 256, 0, stream>>>(attw, attwb, 32768);
  k_cvt<<<32768 / 256, 256, 0, stream>>>(boxw, boxwb, 32768);
  k_cvt<<<262144 / 256, 256, 0, stream>>>(l1w, l1wb, 262144);
  k_cvt<<<262144 / 256, 256, 0, stream>>>(l2w, l2wb, 262144);

  const dim3 blk(256);
  // value projection: v = src @ vpw^T + vpb
  gemm_wmma<256, 256, 0><<<dim3(4, MT / 128), blk, 0, stream>>>(
      srcbf, vpwb, vpb, vbuf, nullptr);
  // attention logits: q @ attw^T + attb
  gemm_wmma<128, 256, 0><<<dim3(2, MT / 128), blk, 0, stream>>>(
      qbf, attwb, attb, awbuf, nullptr);
  // box offsets: q @ boxw^T + boxb
  gemm_wmma<128, 256, 0><<<dim3(2, MT / 128), blk, 0, stream>>>(
      qbf, boxwb, boxb, obbuf, nullptr);
  // softmax + rotated box grid + bilinear sampling -> bf16 attention output
  k_sample<<<MT, blk, 0, stream>>>(vbuf, awbuf, obbuf, rw, outbf);
  // output projection
  gemm_wmma<256, 256, 0><<<dim3(4, MT / 128), blk, 0, stream>>>(
      outbf, opwb, opb, src2, nullptr);
  // LN1(src + src2) -> x1 (f32) + xbf (bf16)
  k_ln<<<MT / 8, blk, 0, stream>>>(src, src2, ln1w, ln1b, x1, xbf, MT);
  // FFN: relu(x @ l1^T + b1) stored bf16, then @ l2^T + b2
  gemm_wmma<1024, 256, 1><<<dim3(16, MT / 128), blk, 0, stream>>>(
      xbf, l1wb, l1b, nullptr, hbf);
  gemm_wmma<256, 1024, 0><<<dim3(4, MT / 128), blk, 0, stream>>>(
      hbf, l2wb, l2b, ffn, nullptr);
  // LN2(x1 + ffn) -> d_out
  k_ln<<<MT / 8, blk, 0, stream>>>(x1, ffn, ln2w, ln2b, (float*)d_out,
                                   nullptr, MT);
}
